// DMRNet_42099269435949
// MI455X (gfx1250) — matrix operand order repaired and checked by reading, not compiled, not run on hardware
//
#include <hip/hip_runtime.h>
#include <math.h>

typedef __attribute__((ext_vector_type(16))) _Float16 v16h;
typedef __attribute__((ext_vector_type(8)))  float    v8f;

#define V_N   256
#define L_N   48
#define M_N   24
#define K_N   10
#define E_N   64
#define MD_N  128
#define MEDV_N 131
#define C_N   300
#define CV_N  300
#define HID2  256      // 2*MD
#define NPAD  144      // MEDV padded to 9 tiles of 16
#define NT_N  9
#define KC_N  8        // 256 / 32

__device__ __forceinline__ float sigmoidf_(float x){ return 1.0f/(1.0f+expf(-x)); }

// ---------------- P1: embedding gather + mean pool + token MLP mean ----------------
__global__ void k_embed(const int* __restrict__ diag, const int* __restrict__ proc,
                        const float* __restrict__ diag_emb, const float* __restrict__ proc_emb,
                        const float* __restrict__ dw1, const float* __restrict__ db1,
                        const float* __restrict__ pw1, const float* __restrict__ pb1,
                        float* __restrict__ patient_rep, float* __restrict__ td_mean,
                        float* __restrict__ tp_mean){
  __shared__ float dt[L_N][E_N];
  __shared__ float pt[L_N][E_N];
  int v = blockIdx.x, t = threadIdx.x;
  for (int idx = t; idx < L_N*E_N; idx += blockDim.x){
    int l = idx >> 6, e = idx & 63;
    dt[l][e] = diag_emb[diag[v*L_N + l]*E_N + e];
    pt[l][e] = proc_emb[proc[v*L_N + l]*E_N + e];
  }
  __syncthreads();
  if (t < E_N){
    float s = 0.f, sp = 0.f;
    for (int l = 0; l < L_N; l++){ s += dt[l][t]; sp += pt[l][t]; }
    patient_rep[v*MD_N + t]       = s  / (float)L_N;
    patient_rep[v*MD_N + E_N + t] = sp / (float)L_N;
  }
  // one thread per hidden column c (blockDim == 128)
  {
    int c = t;
    float ad = 0.f, ap = 0.f;
    for (int l = 0; l < L_N; l++){
      float hd = db1[c], hp = pb1[c];
      for (int e = 0; e < E_N; e++){
        hd += dt[l][e]*dw1[e*MD_N + c];
        hp += pt[l][e]*pw1[e*MD_N + c];
      }
      ad += tanhf(hd); ap += tanhf(hp);
    }
    td_mean[v*MD_N + c] = ad / (float)L_N;
    tp_mean[v*MD_N + c] = ap / (float)L_N;
  }
}

// ---------------- P2: shifted multi-hot histograms ----------------
__global__ void k_multihot(const int* __restrict__ drug_ids, const int* __restrict__ combo_ids,
                           float* __restrict__ drug_mh, float* __restrict__ combo_mh){
  int v = blockIdx.x, t = threadIdx.x;
  if (t < MEDV_N){
    int cnt = 0;
    if (v > 0) for (int m = 0; m < M_N; m++) cnt += (drug_ids[(v-1)*M_N + m] == t+1);
    drug_mh[v*MEDV_N + t] = (float)cnt;
  }
  if (t < C_N){
    int cnt = 0;
    if (v > 0) for (int m = 0; m < K_N; m++) cnt += (combo_ids[(v-1)*K_N + m] == t+1);
    combo_mh[v*C_N + t] = (float)cnt;
  }
}

// ---------------- P3: all_vst_med (depends only on j) + gate weights w = drug + avm ----
__global__ void k_avm(const float* __restrict__ drug_mh, const float* __restrict__ combo_mh,
                      const float* __restrict__ w1, const float* __restrict__ b1,
                      const float* __restrict__ w2, const float* __restrict__ b2,
                      float* __restrict__ wjk){
  __shared__ float x[MEDV_N + C_N];   // 431
  __shared__ float hid[MEDV_N];
  int j = blockIdx.x, t = threadIdx.x;
  for (int idx = t; idx < MEDV_N + C_N; idx += blockDim.x)
    x[idx] = (idx < MEDV_N) ? drug_mh[j*MEDV_N + idx] : combo_mh[j*C_N + (idx - MEDV_N)];
  __syncthreads();
  for (int c = t; c < MEDV_N; c += blockDim.x){
    float s = b1[c];
    for (int q = 0; q < MEDV_N + C_N; q++) s += x[q]*w1[q*MEDV_N + c];
    hid[c] = tanhf(s);
  }
  __syncthreads();
  for (int k = t; k < MEDV_N; k += blockDim.x){
    float s = b2[k];
    for (int c = 0; c < MEDV_N; c++) s += hid[c]*w2[c*MEDV_N + k];
    wjk[j*MEDV_N + k] = tanhf(s) + x[k];     // drug + all_vst_med
  }
}

// ---------------- P4: pr, rank-1 split of pair@pmc_w1, base score, combo head ----------
__global__ void k_dense(const float* __restrict__ patient_rep, const float* __restrict__ td_mean,
                        const float* __restrict__ tp_mean,
                        const float* __restrict__ lw1, const float* __restrict__ lb1,
                        const float* __restrict__ lw2, const float* __restrict__ lb2,
                        const float* __restrict__ pmcw1,
                        const float* __restrict__ med_block,
                        const float* __restrict__ dw2, const float* __restrict__ db2v,
                        const float* __restrict__ pw2, const float* __restrict__ pb2v,
                        const float* __restrict__ dmb, const float* __restrict__ pmb,
                        const float* __restrict__ cw1, const float* __restrict__ cb1,
                        const float* __restrict__ cw2, const float* __restrict__ cb2,
                        float* __restrict__ pr_g, float* __restrict__ Amat,
                        float* __restrict__ Bmat, float* __restrict__ base,
                        float* __restrict__ his_enh, float* __restrict__ combo_out){
  __shared__ float prp[MD_N], hm[MD_N], h1[MD_N], prl[MD_N];
  __shared__ float dh[E_N], ph[E_N], chid[MD_N], td[MD_N], tp[MD_N];
  int v = blockIdx.x, t = threadIdx.x;
  if (t < MD_N){
    prp[t] = patient_rep[v*MD_N + t];
    td[t]  = td_mean[v*MD_N + t];
    tp[t]  = tp_mean[v*MD_N + t];
    hm[t]  = (v > 0) ? patient_rep[(v-1)*MD_N + t] : 0.0f;   // his_mem shift
  }
  __syncthreads();
  if (t < MD_N){               // concat(prp,prp) @ lch_w1 folded
    float s = lb1[t];
    for (int m = 0; m < MD_N; m++) s += prp[m]*(lw1[m*MD_N + t] + lw1[(m+MD_N)*MD_N + t]);
    h1[t] = tanhf(s);
  }
  __syncthreads();
  if (t < MD_N){
    float s = lb2[t];
    for (int m = 0; m < MD_N; m++) s += h1[m]*lw2[m*MD_N + t];
    float pv = tanhf(s);
    prl[t] = pv;
    pr_g[v*MD_N + t] = pv;
  }
  __syncthreads();
  {  // A[i,:] = pr@W1_top ; B[j,:] = his_mem@W1_bot  (blockDim == 256)
    float sa = 0.f, sb = 0.f;
    for (int c = 0; c < MD_N; c++){
      sa += prl[c]*pmcw1[c*HID2 + t];
      sb += hm[c]*pmcw1[(MD_N + c)*HID2 + t];
    }
    Amat[v*HID2 + t] = sa;
    Bmat[v*HID2 + t] = sb;
  }
  if (t < E_N){
    float s = db2v[t], s2 = pb2v[t];
    for (int c = 0; c < MD_N; c++){ s += td[c]*dw2[c*E_N + t]; s2 += tp[c]*pw2[c*E_N + t]; }
    dh[t] = s; ph[t] = s2;
  }
  if (t < MD_N){
    float s = cb1[t];
    for (int m = 0; m < MD_N; m++) s += prp[m]*cw1[m*MD_N + t];
    chid[t] = tanhf(s);
  }
  __syncthreads();
  if (t < MEDV_N){             // base = seq_view(sans his_enh) + token_view
    float s = 0.f;
    for (int c = 0; c < MD_N; c++) s += prl[c]*med_block[c*MEDV_N + t];
    for (int e = 0; e < E_N; e++)  s += dh[e]*dmb[e*MEDV_N + t] + ph[e]*pmb[e*MEDV_N + t];
    base[v*MEDV_N + t] = s;
    his_enh[v*MEDV_N + t] = 0.0f;     // zero for later atomic accumulation
  }
  for (int k = t; k < CV_N; k += blockDim.x){
    if (k == 0){ combo_out[v*CV_N] = 0.0f; }
    else {
      int kk = k - 1;
      float s = cb2[kk];
      for (int c = 0; c < MD_N; c++) s += chid[c]*cw2[c*(CV_N-1) + kk];
      combo_out[v*CV_N + k] = sigmoidf_(s);
    }
  }
}

// ---------------- P5: pack pmc_w2 into per-lane WMMA B-fragments (f16) ----------------
__global__ void k_w2frag(const float* __restrict__ pmcw2, _Float16* __restrict__ frag){
  int b = blockIdx.x;              // b = kc*9 + nt, 72 blocks
  int kc = b / NT_N, nt = b % NT_N;
  int lane = threadIdx.x;          // 32 threads
  for (int h = 0; h < 16; h++){
    int m = kc*32 + ((h >> 3) << 4) + ((lane >> 4) << 3) + (h & 7);   // B 32x16: lanes>=16 hold K+8
    int n = nt*16 + (lane & 15);
    float val = (n < MEDV_N) ? pmcw2[m*MEDV_N + n] : 0.0f;
    frag[((size_t)b*32 + lane)*16 + h] = (_Float16)val;
  }
}

// ---------------- Main WMMA kernel: pairwise container + gated history sum -------------
__global__ void __launch_bounds__(256) k_container(
    const float* __restrict__ Amat, const float* __restrict__ Bmat,
    const float* __restrict__ b1g,  const float* __restrict__ b2g,
    const float* __restrict__ wjk,  const _Float16* __restrict__ w2frag,
    float* __restrict__ his_enh){
  int ti = blockIdx.y, tj = blockIdx.x;
  if (tj > ti) return;                       // tril at tile granularity (uniform exit)
  __shared__ float As[16][HID2];
  __shared__ float Bs[16][HID2];
  __shared__ float b1s[HID2];
  __shared__ float wt[16][NPAD];
  __shared__ float b2s[NPAD];
  int t = threadIdx.x;
  for (int idx = t; idx < 16*HID2; idx += 256){
    int r = idx >> 8, m = idx & 255;
    As[r][m] = Amat[(ti*16 + r)*HID2 + m];
    Bs[r][m] = Bmat[(tj*16 + r)*HID2 + m];
  }
  if (t < HID2) b1s[t] = b1g[t];
  for (int idx = t; idx < 16*NPAD; idx += 256){
    int j = idx / NPAD, k = idx - j*NPAD;
    wt[j][k] = (k < MEDV_N) ? wjk[(tj*16 + j)*MEDV_N + k] : 0.0f;
  }
  if (t < NPAD) b2s[t] = (t < MEDV_N) ? b2g[t] : 0.0f;
  __syncthreads();

  int wave = t >> 5, lane = t & 31;
  int rt0 = wave*2, rt1 = rt0 + 1;           // each wave owns two i rows of the tile
  int jr   = lane & 15;                      // A-frag GEMM row = j_local
  int kgrp = (lane >> 4) << 3;               // A layout: lanes>=16 hold K+8
  v8f acc0[NT_N], acc1[NT_N];
  v8f vz = {};
  #pragma unroll
  for (int n = 0; n < NT_N; n++){ acc0[n] = vz; acc1[n] = vz; }

  const v16h* bfr = (const v16h*)w2frag;
  for (int kc = 0; kc < KC_N; kc++){
    v16h a0, a1;
    #pragma unroll
    for (int h = 0; h < 16; h++){            // 16x32 f16 A-fragment mapping (ISA 7.12.2)
      int m = kc*32 + ((h >> 3) << 4) + kgrp + (h & 7);
      float bb = Bs[jr][m] + b1s[m];
      a0[h] = (_Float16)tanhf(As[rt0][m] + bb);
      a1[h] = (_Float16)tanhf(As[rt1][m] + bb);
    }
    #pragma unroll
    for (int n = 0; n < NT_N; n++){
      v16h bf = bfr[(kc*NT_N + n)*32 + lane];
      acc0[n] = __builtin_amdgcn_wmma_f32_16x16x32_f16(false, a0, false, bf, (short)0, acc0[n], false, false);
      acc1[n] = __builtin_amdgcn_wmma_f32_16x16x32_f16(false, a1, false, bf, (short)0, acc1[n], false, false);
    }
  }

  int colk = lane & 15;
  int rowbase = (lane >> 4) << 3;            // C/D: VGPR v -> row v (+8 for upper lanes)
  #pragma unroll
  for (int rl = 0; rl < 2; rl++){
    int rt = rt0 + rl;
    int ig = ti*16 + rt;
    #pragma unroll
    for (int n = 0; n < NT_N; n++){
      int k = n*16 + colk;
      float s = 0.0f;
      v8f a = rl ? acc1[n] : acc0[n];
      #pragma unroll
      for (int e = 0; e < 8; e++){
        int row = e + rowbase;
        int jg = tj*16 + row;
        float c = tanhf(a[e] + b2s[k]);      // outer tanh -> container
        s += (jg <= ig) ? c*wt[row][k] : 0.0f;   // tril row mask + gate
      }
      float tot = s + __shfl_xor(s, 16, 32); // combine j rows 0-7 with 8-15
      if (lane < 16 && k < MEDV_N)
        atomicAdd(&his_enh[ig*MEDV_N + k], tot);
    }
  }
}

// ---------------- Final: prob, pad, pairwise outer product --------------------------
__global__ void k_prob(const float* __restrict__ base, const float* __restrict__ his_enh,
                       float* __restrict__ out_prob, float* __restrict__ out_pair){
  __shared__ float pl[MEDV_N + 1];
  int v = blockIdx.x, t = threadIdx.x;
  if (t < MEDV_N + 1){
    float p = (t == 0) ? 0.0f : sigmoidf_(base[v*MEDV_N + (t-1)] + his_enh[v*MEDV_N + (t-1)]);
    pl[t] = p;
    out_prob[v*(MEDV_N+1) + t] = p;
  }
  __syncthreads();
  const int W = MEDV_N + 1;                  // 132
  for (int idx = t; idx < W*W; idx += blockDim.x){
    int a = idx / W, b = idx - a*W;
    out_pair[((size_t)a*V_N + v)*W + b] = pl[b]*pl[a];   // pair_out[a,i,b]=prob[i,b]*prob[i,a]
  }
}

extern "C" void kernel_launch(void* const* d_in, const int* in_sizes, int n_in,
                              void* d_out, int out_size, void* d_ws, size_t ws_size,
                              hipStream_t stream){
  (void)in_sizes; (void)n_in; (void)out_size; (void)ws_size;
  const int*   diag      = (const int*)d_in[0];
  const int*   proc      = (const int*)d_in[1];
  const int*   drug_ids  = (const int*)d_in[2];
  const int*   combo_ids = (const int*)d_in[3];
  const float* diag_emb  = (const float*)d_in[4];
  const float* proc_emb  = (const float*)d_in[5];
  const float* dl2_w1=(const float*)d_in[6];  const float* dl2_b1=(const float*)d_in[7];
  const float* dl2_w2=(const float*)d_in[8];  const float* dl2_b2=(const float*)d_in[9];
  const float* pl2_w1=(const float*)d_in[10]; const float* pl2_b1=(const float*)d_in[11];
  const float* pl2_w2=(const float*)d_in[12]; const float* pl2_b2=(const float*)d_in[13];
  const float* med_block=(const float*)d_in[14];
  const float* dmb=(const float*)d_in[15];
  const float* pmb=(const float*)d_in[16];
  const float* pmc_w1=(const float*)d_in[17]; const float* pmc_b1=(const float*)d_in[18];
  const float* pmc_w2=(const float*)d_in[19]; const float* pmc_b2=(const float*)d_in[20];
  const float* clc_w1=(const float*)d_in[21]; const float* clc_b1=(const float*)d_in[22];
  const float* clc_w2=(const float*)d_in[23]; const float* clc_b2=(const float*)d_in[24];
  const float* lch_w1=(const float*)d_in[25]; const float* lch_b1=(const float*)d_in[26];
  const float* lch_w2=(const float*)d_in[27]; const float* lch_b2=(const float*)d_in[28];
  const float* cm_w1=(const float*)d_in[29];  const float* cm_b1=(const float*)d_in[30];
  const float* cm_w2=(const float*)d_in[31];  const float* cm_b2=(const float*)d_in[32];

  float* ws = (float*)d_ws;                  // offsets in floats
  float* patient_rep = ws + 0;               // 256*128
  float* td_mean     = ws + 32768;           // 256*128
  float* tp_mean     = ws + 65536;           // 256*128
  float* drug_mh     = ws + 98304;           // 256*131
  float* combo_mh    = ws + 131840;          // 256*300
  float* wjk         = ws + 208640;          // 256*131
  float* pr_g        = ws + 242176;          // 256*128
  float* Amat        = ws + 274944;          // 256*256
  float* Bmat        = ws + 340480;          // 256*256
  float* base        = ws + 406016;          // 256*131
  float* his_enh     = ws + 439552;          // 256*131
  _Float16* w2frag   = (_Float16*)(ws + 473088);  // 8*9*32*16 halves (32B-aligned)

  float* out       = (float*)d_out;
  float* out_prob  = out;                          // [256,132]
  float* out_pair  = out + 256*132;                // [132,256,132]
  float* out_combo = out + 256*132 + 132*256*132;  // [256,300]

  k_embed   <<<V_N, 128, 0, stream>>>(diag, proc, diag_emb, proc_emb,
                                      dl2_w1, dl2_b1, pl2_w1, pl2_b1,
                                      patient_rep, td_mean, tp_mean);
  k_multihot<<<V_N, 320, 0, stream>>>(drug_ids, combo_ids, drug_mh, combo_mh);
  k_avm     <<<V_N, 256, 0, stream>>>(drug_mh, combo_mh, clc_w1, clc_b1, clc_w2, clc_b2, wjk);
  k_dense   <<<V_N, 256, 0, stream>>>(patient_rep, td_mean, tp_mean,
                                      lch_w1, lch_b1, lch_w2, lch_b2, pmc_w1,
                                      med_block, dl2_w2, dl2_b2, pl2_w2, pl2_b2, dmb, pmb,
                                      cm_w1, cm_b1, cm_w2, cm_b2,
                                      pr_g, Amat, Bmat, base, his_enh, out_combo);
  k_w2frag  <<<KC_N*NT_N, 32, 0, stream>>>(pmc_w2, w2frag);
  k_container<<<dim3(16,16), 256, 0, stream>>>(Amat, Bmat, pmc_b1, pmc_b2, wjk, w2frag, his_enh);
  k_prob    <<<V_N, 256, 0, stream>>>(base, his_enh, out_prob, out_pair);
}
